// MRNN_61452392071437
// MI455X (gfx1250) — matrix operand order repaired and checked by reading, not compiled, not run on hardware
//
#include <hip/hip_runtime.h>
#include <hip/hip_bf16.h>

// MI455X / gfx1250, wave32. Bidirectional per-channel RNN, one
// v_wmma_f32_16x16x32_f16 per step per direction, projection fused into WMMA.
//   A (16x32 f16, constant): M rows 0..7 = [W | V | bias], row 8 = [U | 0 | c0]
//   B (32x16 f16, per step): N=batch, K = [h(0..7) | s(8..10) | 1 | 0]
//   D rows 0..7 = new state (lane n, comps 0..7); D row 8 (df[0], lanes 16..31)
//     = U . h_{t-1} (+c0 fwd) -> stored to LDS by the otherwise-idle high lanes.
// State is kept ONLY as the packed-f16 B operand: per step 4x v_cvt_pk_rtz +
// 4x v_pk_max (ReLU in f16) per direction. A columns K=8..23 are zero, so B lanes
// 16..31 need real data only in slots 8..11 (V|bias|c0): loads unguarded (lanes 16..31 mirror lane&15).

typedef __attribute__((ext_vector_type(16))) _Float16 v16h;
typedef __attribute__((ext_vector_type(8)))  _Float16 v8h;
typedef __attribute__((ext_vector_type(2)))  _Float16 v2h;
typedef __attribute__((ext_vector_type(2)))  __fp16   v2hf;   // cvt_pkrtz result type
typedef __attribute__((ext_vector_type(8)))  float    v8f;

#define Bsz 256
#define Csz 64
#define Tsz 512
#define Hsz 8

union HV16 {
    v16h v;
    _Float16 h[16];
    v2h  p[8];
    v2hf q[8];
    struct LH { v8h lo; v8h hi; } s;
};

__device__ __forceinline__ v8f wmma_step(v16h a, v16h b) {
    v8f cz = {0.f, 0.f, 0.f, 0.f, 0.f, 0.f, 0.f, 0.f};
    // (neg_a, A, neg_b, B, c_mod, C, reuse_a, reuse_b)
    return __builtin_amdgcn_wmma_f32_16x16x32_f16(false, a, false, b,
                                                  (short)0, cz, false, false);
}

__device__ __forceinline__ float relu1(float v) {
    return __builtin_amdgcn_fmed3f(v, 0.f, __builtin_inff());
}

__global__ __launch_bounds__(32)
void mrnn_scan(const float* __restrict__ x,  const float* __restrict__ m,
               const float* __restrict__ dd,
               const float* __restrict__ Wf, const float* __restrict__ Vf,
               const float* __restrict__ cf,
               const float* __restrict__ Wb, const float* __restrict__ Vb,
               const float* __restrict__ cb,
               const float* __restrict__ U,  const float* __restrict__ c0,
               float* __restrict__ xest)
{
    __shared__ float uf_s[Tsz * 16];   // 32 KB: U1.hf[t] + c0  (per batch lane)
    __shared__ float ub_s[Tsz * 16];   // 32 KB: U2.hb, stored time-reversed

    const int lane = threadIdx.x;          // 0..31
    const int b0   = blockIdx.x * 16;      // batch tile base
    const int c    = blockIdx.y;           // channel

    // ---- constant A operands (weights + fused projection row) ----
    HV16 Af, Ab;
    #pragma unroll
    for (int i = 0; i < 16; ++i) { Af.h[i] = (_Float16)0.f; Ab.h[i] = (_Float16)0.f; }
    if (lane < Hsz) {                       // lanes 0..7: rows M=lane, K=0..7 -> W row
        #pragma unroll
        for (int k = 0; k < Hsz; ++k) {
            Af.h[k] = (_Float16)Wf[(c * Hsz + lane) * Hsz + k];
            Ab.h[k] = (_Float16)Wb[(c * Hsz + lane) * Hsz + k];
        }
    } else if (lane == 8) {                 // row M=8, K=0..7 -> projection vectors
        #pragma unroll
        for (int k = 0; k < Hsz; ++k) {
            Af.h[k] = (_Float16)U[c * (2 * Hsz) + k];          // U1
            Ab.h[k] = (_Float16)U[c * (2 * Hsz) + Hsz + k];    // U2
        }
    } else if (lane >= 16 && lane < 16 + Hsz) {  // rows M=lane-16, K=24..27: V + bias
        const int r = lane - 16;
        #pragma unroll
        for (int j = 0; j < 3; ++j) {
            Af.h[8 + j] = (_Float16)Vf[(c * Hsz + r) * 3 + j];
            Ab.h[8 + j] = (_Float16)Vb[(c * Hsz + r) * 3 + j];
        }
        Af.h[11] = (_Float16)cf[c * Hsz + r];
        Ab.h[11] = (_Float16)cb[c * Hsz + r];
    } else if (lane == 24) {                // row M=8, K=27: c0 rides the fwd projection
        Af.h[11] = (_Float16)c0[c];
    }

    // Lanes 16..31 mirror lane&15 (their B slots 8..11 meet V, bias and c0).
    const bool   hi  = (lane >= 16);
    const int    n   = lane & 15;
    const size_t row = ((size_t)(b0 + n) * Csz + c) * (size_t)Tsz;

    // Persistent packed-f16 B operands (state in K=0..7, inputs in K=8..11).
    HV16 Bf, Bb;
    #pragma unroll
    for (int i = 0; i < 16; ++i) { Bf.h[i] = (_Float16)0.f; Bb.h[i] = (_Float16)0.f; }
    const v8h zh = { (_Float16)0.f, (_Float16)0.f, (_Float16)0.f, (_Float16)0.f,
                     (_Float16)0.f, (_Float16)0.f, (_Float16)0.f, (_Float16)0.f };

    // T+1 iterations: iteration t emits state h_t (rows 0..7) and the fused
    // projection of h_{t-1} (row 8). Store guard t>0; iteration T only projects.
    for (int t = 0; t <= Tsz; ++t) {
        // edge-padded indices: in_f[t] = x[clamp(t-1)], in_b[t] = x[clamp(T-t)]
        const int itf = (t == 0) ? 0 : (t - 1);
        const int itb = (t == 0) ? (Tsz - 1) : (Tsz - t);

        const float sfx = x [row + itf];
        const float sfm = m [row + itf];
        const float sfd = dd[row + itf];
        const float sbx = x [row + itb];
        const float sbm = m [row + itb];
        const float sbd = dd[row + itb];

        // branch-free speculative prefetch of both streaming directions
        __builtin_prefetch(&x [row + itf + 64], 0, 1);
        __builtin_prefetch(&m [row + itf + 64], 0, 1);
        __builtin_prefetch(&dd[row + itf + 64], 0, 1);
        __builtin_prefetch(&x [row + itb - 64], 0, 1);
        __builtin_prefetch(&m [row + itb - 64], 0, 1);
        __builtin_prefetch(&dd[row + itb - 64], 0, 1);

        Bf.q[4] = __builtin_amdgcn_cvt_pkrtz(sfx, sfm);  // K=8,9
        Bb.q[4] = __builtin_amdgcn_cvt_pkrtz(sbx, sbm);
        Bf.q[5] = __builtin_amdgcn_cvt_pkrtz(sfd, 1.f);  // K=10, bias column K=27 (slot 11, lanes 16..31)
        Bb.q[5] = __builtin_amdgcn_cvt_pkrtz(sbd, 1.f);

        // ---- two independent WMMA chains (fwd + bwd) ----
        const v8f df = wmma_step(Af.v, Bf.v);
        const v8f db = wmma_step(Ab.v, Bb.v);

        // Fused projections of h_{t-1}: D row 8 = df[0]/db[0] on lanes 16..31.
        if (t > 0 && hi) {
            uf_s[(t - 1) * 16 + n]   = df[0];   // U1.hf[t-1] + c0
            ub_s[(Tsz - t) * 16 + n] = db[0];   // U2.hb[t-1], time-reversed slot
        }

        // ---- state update entirely in packed f16: pack then ReLU ----
        Bf.q[0] = __builtin_amdgcn_cvt_pkrtz(df[0], df[1]);
        Bf.q[1] = __builtin_amdgcn_cvt_pkrtz(df[2], df[3]);
        Bf.q[2] = __builtin_amdgcn_cvt_pkrtz(df[4], df[5]);
        Bf.q[3] = __builtin_amdgcn_cvt_pkrtz(df[6], df[7]);
        Bb.q[0] = __builtin_amdgcn_cvt_pkrtz(db[0], db[1]);
        Bb.q[1] = __builtin_amdgcn_cvt_pkrtz(db[2], db[3]);
        Bb.q[2] = __builtin_amdgcn_cvt_pkrtz(db[4], db[5]);
        Bb.q[3] = __builtin_amdgcn_cvt_pkrtz(db[6], db[7]);
        Bf.s.lo = __builtin_elementwise_max(Bf.s.lo, zh);   // v_pk_max_num_f16
        Bb.s.lo = __builtin_elementwise_max(Bb.s.lo, zh);
    }

    // ---- fused x_est = relu(uf + ub), coalesced over t (c0 already in uf) ----
    for (int i = lane; i < 16 * Tsz; i += 32) {
        const int nn = i >> 9;             // batch within tile
        const int tt = i & (Tsz - 1);      // time
        const float xe = relu1(uf_s[tt * 16 + nn] + ub_s[tt * 16 + nn]);
        xest[((size_t)(b0 + nn) * Csz + c) * (size_t)Tsz + tt] = xe;
    }
}

// Tiny 64 -> 3 -> 64 channel mix per (b, t). Bandwidth bound; fully coalesced.
__global__ __launch_bounds__(256)
void mrnn_mix(const float* __restrict__ x,   const float* __restrict__ m,
              const float* __restrict__ xest,
              const float* __restrict__ V1w, const float* __restrict__ V2w,
              const float* __restrict__ Uw,  const float* __restrict__ Ubv,
              const float* __restrict__ Ww,  const float* __restrict__ Wbv,
              float* __restrict__ out)
{
    __shared__ float sV1[192], sV2[192], sU[192], sW[192], sWb[64], sUb[3];
    const int tid = threadIdx.x;
    if (tid < 192) {
        sV1[tid] = V1w[tid];
        sV2[tid] = V2w[tid];
        const int i = tid >> 6, cc = tid & 63;
        sU[tid] = (cc == i) ? 0.f : Uw[tid];   // zero diagonal (rows 0..2)
        sW[tid] = Ww[tid];
    }
    if (tid < 64) sWb[tid] = Wbv[tid];
    if (tid < 3)  sUb[tid] = Ubv[tid];
    __syncthreads();

    const int p = blockIdx.x * 256 + tid;      // (b, t) position
    const int b = p >> 9;
    const int t = p & (Tsz - 1);
    const size_t base = (size_t)b * Csz * Tsz + t;

    float a0 = sUb[0], a1 = sUb[1], a2 = sUb[2];
    #pragma unroll 4
    for (int cc = 0; cc < Csz; ++cc) {
        const size_t idx = base + (size_t)cc * Tsz;
        const float xe = xest[idx];
        const float mm = m[idx];
        const float xx = x[idx];
        a0 += sV1[cc]       * xe + sV2[cc]       * mm + sU[cc]       * xx;
        a1 += sV1[64 + cc]  * xe + sV2[64 + cc]  * mm + sU[64 + cc]  * xx;
        a2 += sV1[128 + cc] * xe + sV2[128 + cc] * mm + sU[128 + cc] * xx;
    }
    const float h0 = relu1(a0);
    const float h1 = relu1(a1);
    const float h2 = relu1(a2);
    #pragma unroll 4
    for (int cc = 0; cc < Csz; ++cc) {
        out[base + (size_t)cc * Tsz] =
            sW[cc * 3] * h0 + sW[cc * 3 + 1] * h1 + sW[cc * 3 + 2] * h2 + sWb[cc];
    }
}

extern "C" void kernel_launch(void* const* d_in, const int* in_sizes, int n_in,
                              void* d_out, int out_size, void* d_ws, size_t ws_size,
                              hipStream_t stream) {
    const float* x    = (const float*)d_in[0];
    const float* m    = (const float*)d_in[1];
    const float* dd   = (const float*)d_in[2];
    const float* Wf   = (const float*)d_in[3];
    const float* Vf   = (const float*)d_in[4];
    const float* cf   = (const float*)d_in[5];
    const float* Wb   = (const float*)d_in[6];
    const float* Vb   = (const float*)d_in[7];
    const float* cb   = (const float*)d_in[8];
    const float* U    = (const float*)d_in[9];
    const float* c0   = (const float*)d_in[10];
    const float* V1w  = (const float*)d_in[11];
    const float* V2w  = (const float*)d_in[12];
    const float* Uw   = (const float*)d_in[13];
    const float* Ubv  = (const float*)d_in[14];
    const float* Ww   = (const float*)d_in[15];
    const float* Wbv  = (const float*)d_in[16];
    float* out  = (float*)d_out;
    float* xest = (float*)d_ws;            // B*C*T floats = 32 MiB scratch

    // 64 channels x 16 batch-tiles, one wave32 each: both scans + x_est fused.
    mrnn_scan<<<dim3(16, Csz), 32, 0, stream>>>(x, m, dd, Wf, Vf, cf,
                                                Wb, Vb, cb, U, c0, xest);
    // (B*T)/256 = 512 blocks of 256 threads.
    mrnn_mix<<<(Bsz * Tsz) / 256, 256, 0, stream>>>(x, m, xest, V1w, V2w,
                                                    Uw, Ubv, Ww, Wbv, out);
}
